// UniCrystalFormer_59167469470431
// MI455X (gfx1250) — compile-verified, matrix operand order checked by reading, not run on hardware
//
#include <hip/hip_runtime.h>
#include <hip/hip_bf16.h>
#include <math.h>

// ---------------- problem constants ----------------
#define N_NODES  8000
#define N_EDGES  64000
#define N_GRAPHS 256
#define HID      128
#define HEADS    4
#define HC       512           // HEADS*HID
#define H3       384           // 3*HID
#define EH       (N_EDGES*HEADS)
#define RES_SCALE 0.5f

typedef _Float16 h16;
typedef __attribute__((ext_vector_type(16))) _Float16 v16h;
typedef __attribute__((ext_vector_type(8)))  _Float16 v8h;
typedef __attribute__((ext_vector_type(8)))  float    v8f;
typedef __attribute__((ext_vector_type(4)))  unsigned int u32x4;
typedef __attribute__((ext_vector_type(8)))  int          i32x8;
typedef __attribute__((ext_vector_type(4)))  int          i32x4;

union Frag { v16h v; struct { v8h lo, hi; } h; };

static __device__ __forceinline__ float sigm(float x){ return 1.f/(1.f+__expf(-x)); }

// TDM (Tensor Data Mover) availability — guarded so either toolchain compiles.
#if defined(__AMDGCN__) && __has_builtin(__builtin_amdgcn_tensor_load_to_lds) && \
    __has_builtin(__builtin_amdgcn_s_wait_tensorcnt)
#define USE_TDM 1
#else
#define USE_TDM 0
#endif

// =====================================================================
// Universal WMMA GEMM:  out[M,N] = epi( A[M,K] @ W[K,N] + bias )
//  * A addressed via up to 3 K-segments of 128: A(r,k) = baseS[offS[r]+(k&127)],
//    S = k>>7  (covers plain GEMMs and the concat-gather GEMMs).
//  * Weights pre-converted to f16 and pre-TRANSPOSED: Wt[N,K].
//    Each 32xK-chunk B tile (128 rows of N, 32 contiguous K) is a uniform 2D
//    tile -> fed by one TENSOR_LOAD_TO_LDS per chunk (wave 0 issues, all waves
//    consume after s_wait_tensorcnt + barrier). Fallback: coalesced v8h copies.
//  * Block = 256 threads = 8 wave32, tile 128x128, K stepped by 32 with
//    v_wmma_f32_16x16x32_f16 (f16 in, f32 accumulate).
//  ACT: 0=none 1=silu 2=softplus 3=sigmoid ; optional elementwise mul.
// =====================================================================
#define LDSROWA 40   // A: 32 f16 data + 8 pad -> 80B stride, 16B-aligned frags
#define LDSROWB 32   // B: packed rows (matches TDM contiguous tile layout)

template<int ACT>
__global__ __launch_bounds__(256)
void k_gemm(const float* __restrict__ b0, const float* __restrict__ b1,
            const float* __restrict__ b2,
            const int* __restrict__ o0, const int* __restrict__ o1,
            const int* __restrict__ o2,
            const h16* __restrict__ Wt,     // [N,K] f16 (pre-transposed)
            const float* __restrict__ bias,
            const float* __restrict__ mul, float* __restrict__ out,
            int M, int K, int N)
{
  __shared__ __align__(16) h16 As[128*LDSROWA];
  __shared__ __align__(16) h16 Bs[128*LDSROWB];

  const int tid  = threadIdx.x;
  const int wave = tid >> 5;
  const int lane = tid & 31;
  const int half = lane >> 4;
  const int lrow = lane & 15;
  const int m0 = blockIdx.x * 128;
  const int n0 = blockIdx.y * 128;

  v8f acc[8];
  for (int i = 0; i < 8; ++i)
    for (int j = 0; j < 8; ++j) acc[i][j] = 0.f;

  // A staging: 2 threads per row, 16 consecutive k each (f32 -> f16 convert)
  const int arow  = tid >> 1;
  const int akoff = (tid & 1) * 16;
  int r = m0 + arow; if (r >= M) r = M - 1;

  for (int kc = 0; kc < K; kc += 32) {
#if USE_TDM
    // ---- issue DMA of the B tile: Wt[n0..n0+127][kc..kc+31] -> Bs ----
    if (wave == 0) {
      unsigned long long ga =
          (unsigned long long)(const void*)(Wt + (size_t)n0 * K + kc);
      unsigned lds_off = (unsigned)(unsigned long long)(const void*)&Bs[0];
      u32x4 g0 = {0u, 0u, 0u, 0u};
      i32x8 g1 = {0, 0, 0, 0, 0, 0, 0, 0};
      i32x4 g2 = {0, 0, 0, 0};
      i32x4 g3 = {0, 0, 0, 0};
      g0[0] = 1u;                                   // count=1, no gather
      g0[1] = lds_off;                              // lds_addr (bytes)
      g0[2] = (unsigned)ga;                         // global_addr[31:0]
      g0[3] = (unsigned)((ga >> 32) & 0x1FFFFFFu)   // global_addr[56:32]
              | (2u << 30);                         // type=2 (image)
      g1[0] = (int)(1u << 16);                      // data_size=1 -> 2 bytes
      g1[1] = (int)(32u << 16);                     // tensor_dim0 = 32 (lo16)
      g1[2] = (int)(128u << 16);                    // dim0 hi=0 | tensor_dim1=128
      g1[3] = (int)(32u << 16);                     // dim1 hi=0 | tile_dim0=32
      g1[4] = 128;                                  // tile_dim1=128, tile_dim2=0
      g1[5] = K;                                    // tensor_dim0_stride lo32
      g1[6] = 0;                                    // stride hi | dim1_stride lo
      g1[7] = 0;
#if __clang_major__ >= 23
      i32x8 g4 = {0, 0, 0, 0, 0, 0, 0, 0};
      __builtin_amdgcn_tensor_load_to_lds(g0, g1, g2, g3, g4, 0);
#else
      __builtin_amdgcn_tensor_load_to_lds(g0, g1, g2, g3, 0);
#endif
    }
    if (wave == 1 && kc + 32 < K)                   // warm GL2 for next chunk
      __builtin_prefetch(Wt + (size_t)(n0 + lane * 4) * K + kc + 32, 0, 3);
#else
    // ---- fallback: coalesced f16 copy (Wt already [N,K]) ----
    {
      const int brow  = tid >> 1;
      const int bkoff = (tid & 1) * 16;
      const h16* wp = Wt + (size_t)(n0 + brow) * K + kc + bkoff;
      v8h w0 = *(const v8h*)wp;
      v8h w1 = *(const v8h*)(wp + 8);
      *(v8h*)&Bs[brow * LDSROWB + bkoff]     = w0;
      *(v8h*)&Bs[brow * LDSROWB + bkoff + 8] = w1;
      if (kc + 32 < K)
        __builtin_prefetch(wp + 32, 0, 3);
    }
#endif

    // ---- A staging overlaps the B-tile DMA ----
    {
      const int seg = kc >> 7;
      const float* ab = (seg == 0) ? b0 : ((seg == 1) ? b1 : b2);
      const int*   ao = (seg == 0) ? o0 : ((seg == 1) ? o1 : o2);
      const float* ap = ab + ao[r] + (kc & 127) + akoff;
      #pragma unroll
      for (int j = 0; j < 16; ++j)
        As[arow * LDSROWA + akoff + j] = (h16)ap[j];
    }

#if USE_TDM
    if (wave == 0) __builtin_amdgcn_s_wait_tensorcnt(0);
#endif
    __syncthreads();

    // A fragment: lane<16 -> K {0..7,16..23}, lane>=16 -> K {8..15,24..31}
    Frag a;
    a.h.lo = *(const v8h*)&As[(wave * 16 + lrow) * LDSROWA + half * 8];
    a.h.hi = *(const v8h*)&As[(wave * 16 + lrow) * LDSROWA + 16 + half * 8];

    #pragma unroll
    for (int nt = 0; nt < 8; ++nt) {
      Frag b;
      b.h.lo = *(const v8h*)&Bs[(nt * 16 + lrow) * LDSROWB + half * 8];
      b.h.hi = *(const v8h*)&Bs[(nt * 16 + lrow) * LDSROWB + 16 + half * 8];
      acc[nt] = __builtin_amdgcn_wmma_f32_16x16x32_f16(
          false, a.v, false, b.v, (short)0, acc[nt], false, false);
    }
    __syncthreads();
  }

  // Epilogue. C/D layout: lane<16: row=v, lane>=16: row=8+v ; col=lane&15.
  #pragma unroll
  for (int nt = 0; nt < 8; ++nt) {
    const int c = n0 + nt * 16 + lrow;
    const float bia = bias ? bias[c] : 0.f;
    #pragma unroll
    for (int v = 0; v < 8; ++v) {
      const int rr = m0 + wave * 16 + half * 8 + v;
      if (rr < M) {
        float x = acc[nt][v] + bia;
        if (ACT == 1) x = x * sigm(x);                        // silu
        if (ACT == 2) x = (x > 20.f) ? x : log1pf(__expf(x)); // softplus
        if (ACT == 3) x = sigm(x);                            // sigmoid
        if (mul) x *= mul[(size_t)rr * N + c];
        out[(size_t)rr * N + c] = x;
      }
    }
  }
}

// ---------------- weight convert+transpose: f32[K,N] -> f16[N,K] ----------------
__global__ void k_cvtT(const float* __restrict__ W, h16* __restrict__ Wt, int K, int N) {
  for (int i = blockIdx.x * blockDim.x + threadIdx.x; i < K * N;
       i += gridDim.x * blockDim.x) {
    int k = i / N, n = i - k * N;
    Wt[(size_t)n * K + k] = (h16)W[i];
  }
}

// ---------------- small utility kernels ----------------
__global__ void k_zero(float* p, int n) {
  for (int i = blockIdx.x * blockDim.x + threadIdx.x; i < n; i += gridDim.x * blockDim.x)
    p[i] = 0.f;
}

__global__ void k_iota(int* o, int n, int scale, int add) {
  for (int i = blockIdx.x * blockDim.x + threadIdx.x; i < n; i += gridDim.x * blockDim.x)
    o[i] = i * scale + add;
}

__global__ void k_gath_off(const int* __restrict__ idx, int* o, int n, int scale) {
  for (int i = blockIdx.x * blockDim.x + threadIdx.x; i < n; i += gridDim.x * blockDim.x)
    o[i] = idx[i] * scale;
}

__global__ void k_matoff(const int* __restrict__ dst, const int* __restrict__ src,
                         int* o0, int* o1, int* o2) {
  for (int i = blockIdx.x * blockDim.x + threadIdx.x; i < EH; i += gridDim.x * blockDim.x) {
    int e = i >> 2, h = i & 3;
    o0[i] = dst[e] * HC + h * HID;
    o1[i] = src[e] * HC + h * HID;
    o2[i] = e * HC + h * HID;
  }
}

__global__ void k_embed(const int* __restrict__ t, const float* __restrict__ emb,
                        float* xc, float* xm) {
  for (int i = blockIdx.x * blockDim.x + threadIdx.x; i < N_NODES * HID;
       i += gridDim.x * blockDim.x) {
    float v = emb[t[i >> 7] * HID + (i & 127)];
    xc[i] = v; xm[i] = v;
  }
}

__global__ void k_rbf(const float* __restrict__ ev, float* rbf) {
  const float step = 8.0f / 127.0f;
  const float gamma = 1.0f / (step * step);
  for (int i = blockIdx.x * blockDim.x + threadIdx.x; i < N_EDGES * HID;
       i += gridDim.x * blockDim.x) {
    int e = i >> 7, j = i & 127;
    float x = ev[e * 3], y = ev[e * 3 + 1], z = ev[e * 3 + 2];
    float d = sqrtf(x * x + y * y + z * z) - (float)j * step;
    rbf[i] = __expf(-gamma * d * d);
  }
}

// column stats (sum, sumsq) over [M,128] -> st[0..127]=sum, st[128..255]=sumsq
__global__ void k_colstats(const float* __restrict__ x, int M, float* st) {
  int c = threadIdx.x & 127;
  int r = blockIdx.x * 2 + (threadIdx.x >> 7);
  float s = 0.f, q = 0.f;
  for (; r < M; r += gridDim.x * 2) {
    float v = x[(size_t)r * HID + c];
    s += v; q += v * v;
  }
  atomicAdd(&st[c], s);
  atomicAdd(&st[HID + c], q);
}

__global__ void k_bnsig(const float* __restrict__ h, const float* __restrict__ st,
                        const float* __restrict__ g, const float* __restrict__ b,
                        float* sigma, float* ec, int M) {
  float invM = 1.f / (float)M;
  for (int i = blockIdx.x * blockDim.x + threadIdx.x; i < M * HID;
       i += gridDim.x * blockDim.x) {
    int c = i & 127;
    float mean = st[c] * invM;
    float var  = st[HID + c] * invM - mean * mean;
    float s = sigm((h[i] - mean) * rsqrtf(var + 1e-5f) * g[c] + b[c]);
    sigma[i] = s;
    ec[i] += s;
  }
}

__global__ void k_bnsilu_add(const float* __restrict__ agg, const float* __restrict__ st,
                             const float* __restrict__ g, const float* __restrict__ b,
                             const float* __restrict__ xin, float* xout, int M) {
  float invM = 1.f / (float)M;
  for (int i = blockIdx.x * blockDim.x + threadIdx.x; i < M * HID;
       i += gridDim.x * blockDim.x) {
    int c = i & 127;
    float mean = st[c] * invM;
    float var  = st[HID + c] * invM - mean * mean;
    float x = (agg[i] - mean) * rsqrtf(var + 1e-5f) * g[c] + b[c];
    x = x * sigm(x);
    xout[i] = x + xin[i];
  }
}

__global__ void k_scatter_add(const float* __restrict__ v, const int* __restrict__ dst,
                              float* o, int E) {
  for (int i = blockIdx.x * blockDim.x + threadIdx.x; i < E * HID;
       i += gridDim.x * blockDim.x)
    atomicAdd(&o[dst[i >> 7] * HID + (i & 127)], v[i]);
}

__global__ void k_scatter_heads(const float* __restrict__ m, const int* __restrict__ dst,
                                float* o) {
  for (int i = blockIdx.x * blockDim.x + threadIdx.x; i < EH * HID;
       i += gridDim.x * blockDim.x) {
    int r = i >> 7, e = r >> 2, c = i & 127;
    atomicAdd(&o[dst[e] * HID + c], 0.25f * m[i]);   // head-mean fused into segsum
  }
}

// ----- GraphNorm / segment mean -----
__global__ void k_segsum(const float* __restrict__ x, const int* __restrict__ batch,
                         float* gs, float* gc, int n) {
  for (int i = blockIdx.x * blockDim.x + threadIdx.x; i < n * HID;
       i += gridDim.x * blockDim.x) {
    int r = i >> 7, c = i & 127;
    int g = batch[r];
    atomicAdd(&gs[g * HID + c], x[i]);
    if (c == 0) atomicAdd(&gc[g], 1.f);
  }
}

__global__ void k_gn_center(const float* __restrict__ x, const int* __restrict__ batch,
                            const float* __restrict__ gs, const float* __restrict__ gc,
                            const float* __restrict__ ms, float* tmp, float* gvar, int n) {
  for (int i = blockIdx.x * blockDim.x + threadIdx.x; i < n * HID;
       i += gridDim.x * blockDim.x) {
    int r = i >> 7, c = i & 127;
    int g = batch[r];
    float mean = gs[g * HID + c] / fmaxf(gc[g], 1.f);
    float t = x[i] - mean * ms[c];
    tmp[i] = t;
    atomicAdd(&gvar[g * HID + c], t * t);
  }
}

__global__ void k_gn_final(const float* __restrict__ tmp, const int* __restrict__ batch,
                           const float* __restrict__ gvar, const float* __restrict__ gc,
                           const float* __restrict__ w, const float* __restrict__ b,
                           float* out, int n) {
  for (int i = blockIdx.x * blockDim.x + threadIdx.x; i < n * HID;
       i += gridDim.x * blockDim.x) {
    int r = i >> 7, c = i & 127;
    int g = batch[r];
    float var = gvar[g * HID + c] / fmaxf(gc[g], 1.f);
    out[i] = w[c] * tmp[i] * rsqrtf(var + 1e-5f) + b[c];
  }
}

__global__ void k_segdiv(const float* __restrict__ gs, const float* __restrict__ gc,
                         float* feat) {
  for (int i = blockIdx.x * blockDim.x + threadIdx.x; i < N_GRAPHS * HID;
       i += gridDim.x * blockDim.x)
    feat[i] = gs[i] / fmaxf(gc[i >> 7], 1.f);
}

// ----- matformer attention gate: sigmoid(LN(q_cat * key_cat / sqrt(3C))) -----
__global__ __launch_bounds__(128)
void k_gate(const float* __restrict__ q, const float* __restrict__ kk,
            const float* __restrict__ ep, const int* __restrict__ dst,
            const int* __restrict__ src, const float* __restrict__ g,
            const float* __restrict__ b, float* G) {
  const int rrow = blockIdx.x;           // 0..EH-1
  const int e = rrow >> 2, h = rrow & 3;
  const int t = threadIdx.x;             // 0..127
  const int d = dst[e], s = src[e];
  const float sc = 0.051031036307982884f;   // 1/sqrt(384)
  float qv = q[d * HC + h * HID + t];
  float a0 = qv * kk[d * HC + h * HID + t] * sc;
  float a1 = qv * kk[s * HC + h * HID + t] * sc;
  float a2 = qv * ep[(size_t)e * HC + h * HID + t] * sc;
  __shared__ float red[128];
  red[t] = a0 + a1 + a2;
  __syncthreads();
  for (int o = 64; o > 0; o >>= 1) { if (t < o) red[t] += red[t + o]; __syncthreads(); }
  float mean = red[0] / (float)H3;
  __syncthreads();
  float d0 = a0 - mean, d1 = a1 - mean, d2 = a2 - mean;
  red[t] = d0 * d0 + d1 * d1 + d2 * d2;
  __syncthreads();
  for (int o = 64; o > 0; o >>= 1) { if (t < o) red[t] += red[t + o]; __syncthreads(); }
  float rstd = rsqrtf(red[0] / (float)H3 + 1e-5f);
  size_t base = (size_t)rrow * H3;
  G[base + t]           = sigm(d0 * rstd * g[t] + b[t]);
  G[base + HID + t]     = sigm(d1 * rstd * g[HID + t] + b[HID + t]);
  G[base + 2 * HID + t] = sigm(d2 * rstd * g[2 * HID + t] + b[2 * HID + t]);
}

// LayerNorm over last 128 dims, in-place
__global__ __launch_bounds__(128)
void k_ln128(float* x, const float* __restrict__ g, const float* __restrict__ b) {
  const int r = blockIdx.x, t = threadIdx.x;
  float v = x[(size_t)r * HID + t];
  __shared__ float red[128];
  red[t] = v;
  __syncthreads();
  for (int o = 64; o > 0; o >>= 1) { if (t < o) red[t] += red[t + o]; __syncthreads(); }
  float mean = red[0] / (float)HID;
  __syncthreads();
  float dv = v - mean;
  red[t] = dv * dv;
  __syncthreads();
  for (int o = 64; o > 0; o >>= 1) { if (t < o) red[t] += red[t + o]; __syncthreads(); }
  float rstd = rsqrtf(red[0] / (float)HID + 1e-5f);
  x[(size_t)r * HID + t] = dv * rstd * g[t] + b[t];
}

// beta = sigmoid([out, x_r, out-x_r] @ bw) ; xm = beta*x_r + (1-beta)*out
__global__ __launch_bounds__(128)
void k_beta_mix(const float* __restrict__ matt, const float* __restrict__ xr,
                const float* __restrict__ bw, float* xm) {
  const int i = blockIdx.x, t = threadIdx.x;
  float o = matt[i * HID + t], r = xr[i * HID + t];
  __shared__ float red[128];
  red[t] = o * bw[t] + r * bw[HID + t] + (o - r) * bw[2 * HID + t];
  __syncthreads();
  for (int s = 64; s > 0; s >>= 1) { if (t < s) red[t] += red[t + s]; __syncthreads(); }
  float beta = sigm(red[0]);
  xm[i * HID + t] = beta * r + (1.f - beta) * o;
}

// residual gate mixer update
__global__ void k_mix_update(const float* __restrict__ gate, const float* __restrict__ xc,
                             const float* __restrict__ xm, float* x_c, float* x_m) {
  for (int i = blockIdx.x * blockDim.x + threadIdx.x; i < N_NODES * HID;
       i += gridDim.x * blockDim.x) {
    float g = gate[i];
    float xo = g * xc[i] + (1.f - g) * xm[i];
    x_c[i] = xc[i] + RES_SCALE * xo;
    x_m[i] = xm[i] + RES_SCALE * xo;
  }
}

__global__ void k_avg2(const float* a, const float* b, float* o) {
  for (int i = blockIdx.x * blockDim.x + threadIdx.x; i < N_NODES * HID;
       i += gridDim.x * blockDim.x)
    o[i] = 0.5f * (a[i] + b[i]);
}

__global__ __launch_bounds__(128)
void k_fc2(const float* __restrict__ fh, const float* __restrict__ w,
           const float* __restrict__ b, float* out) {
  const int g = blockIdx.x, t = threadIdx.x;
  __shared__ float red[128];
  red[t] = fh[g * HID + t] * w[t];
  __syncthreads();
  for (int o = 64; o > 0; o >>= 1) { if (t < o) red[t] += red[t + o]; __syncthreads(); }
  if (t == 0) out[g] = red[0] + b[0];
}

// =====================================================================
// host side
// =====================================================================
struct LayerP {
  const float *cg1w,*cg1b,*cg2w,*cg2b,*ca1w,*ca1b,*ca2w,*ca2b;
  const float *bneg,*bneb,*bnng,*bnnb;
  const float *qw,*qb,*kw,*kb,*vw,*vb,*ew,*eb,*skw,*skb;
  const float *betaw;
  const float *u1w,*u1b,*u2w,*u2b,*msgw,*msgb;
  const float *lnag,*lnab,*lnmg,*lnmb;
  const float *gncw,*gncb,*gncm,*gnmw,*gnmb,*gnmm;
  const float *mixw,*mixb;
};
struct LayerW {   // f16 pre-transposed weights [N,K]
  h16 *cg1,*cg2,*ca1,*ca2,*q,*k,*v,*e,*skip,*u1,*u2,*msg,*mix;
};

static inline int cdiv(int a, int b) { return (a + b - 1) / b; }

extern "C" void kernel_launch(void* const* d_in, const int* in_sizes, int n_in,
                              void* d_out, int out_size, void* d_ws, size_t ws_size,
                              hipStream_t stream) {
  (void)in_sizes; (void)n_in; (void)out_size; (void)ws_size;
  const int*   atom_types = (const int*)d_in[0];
  const float* edge_vec   = (const float*)d_in[1];
  const int*   ei         = (const int*)d_in[2];
  const int*   batch      = (const int*)d_in[3];
  const int* src = ei;              // edge_index[0]
  const int* dst = ei + N_EDGES;    // edge_index[1]

  // ---- walk params (pytree leaves in source/insertion order) ----
  int cur = 4;
  auto nx = [&]() { return (const float*)d_in[cur++]; };
  const float* emb  = nx();
  const float *rbf1w = nx(), *rbf1b = nx(), *rbf2w = nx(), *rbf2b = nx();
  const float *fc1w  = nx(), *fc1b  = nx(), *fc2w  = nx(), *fc2b  = nx();
  LayerP L[5];
  for (int l = 0; l < 5; ++l) {
    LayerP& p = L[l];
    p.cg1w=nx(); p.cg1b=nx(); p.cg2w=nx(); p.cg2b=nx();
    p.ca1w=nx(); p.ca1b=nx(); p.ca2w=nx(); p.ca2b=nx();
    p.bneg=nx(); p.bneb=nx(); p.bnng=nx(); p.bnnb=nx();
    p.qw=nx(); p.qb=nx(); p.kw=nx(); p.kb=nx(); p.vw=nx(); p.vb=nx();
    p.ew=nx(); p.eb=nx(); p.skw=nx(); p.skb=nx();
    p.betaw=nx();
    p.u1w=nx(); p.u1b=nx(); p.u2w=nx(); p.u2b=nx(); p.msgw=nx(); p.msgb=nx();
    p.lnag=nx(); p.lnab=nx(); p.lnmg=nx(); p.lnmb=nx();
    p.gncw=nx(); p.gncb=nx(); p.gncm=nx();
    p.gnmw=nx(); p.gnmb=nx(); p.gnmm=nx();
    p.mixw=nx(); p.mixb=nx();
  }

  // ---- workspace bump allocator ----
  char* wp = (char*)d_ws;
  auto allocF = [&](size_t n) {
    float* p = (float*)wp; wp += ((n * 4) + 255) & ~(size_t)255; return p;
  };
  auto allocI = [&](size_t n) {
    int* p = (int*)wp; wp += ((n * 4) + 255) & ~(size_t)255; return p;
  };
  auto allocH = [&](size_t n) {
    h16* p = (h16*)wp; wp += ((n * 2) + 255) & ~(size_t)255; return p;
  };
  const size_t NODE = (size_t)N_NODES * HID;
  const size_t EDGE = (size_t)N_EDGES * HID;
  float *x_c = allocF(NODE), *x_m = allocF(NODE), *xc = allocF(NODE), *xm = allocF(NODE);
  float *x_r = allocF(NODE), *agg = allocF(NODE), *tmpn = allocF(NODE);
  float *gateN = allocF(NODE), *matt = allocF(NODE);
  float *qb = allocF((size_t)N_NODES * HC), *kb = allocF((size_t)N_NODES * HC);
  float *vb = allocF((size_t)N_NODES * HC);
  float *ef = allocF(EDGE), *e_c = allocF(EDGE), *sigma = allocF(EDGE);
  float *hg = allocF(EDGE), *he = allocF(EDGE);
  float *epb  = allocF((size_t)N_EDGES * HC);
  float *G    = allocF((size_t)EH * H3);
  float *U1   = allocF((size_t)EH * H3);
  float *mraw = allocF((size_t)EH * HID);
  float *gsum = allocF(N_GRAPHS * HID), *gcnt = allocF(N_GRAPHS);
  float *gvar = allocF(N_GRAPHS * HID), *feat = allocF(N_GRAPHS * HID);
  float *feath = allocF(N_GRAPHS * HID);
  float *st = allocF(2 * HID);
  int *offA128 = allocI(EH);                  // r*128 (any 128-stride buffer)
  int *offP0 = allocI(EH), *offP1 = allocI(EH), *offP2 = allocI(EH); // r*384+{0,128,256}
  int *cartD = allocI(N_EDGES), *cartS = allocI(N_EDGES);
  int *matO0 = allocI(EH), *matO1 = allocI(EH), *matO2 = allocI(EH);
  // f16 transposed weights
  h16 *rbf1t = allocH(128*128), *rbf2t = allocH(128*128), *fc1t = allocH(128*128);
  LayerW W[5];
  for (int l = 0; l < 5; ++l) {
    W[l].cg1 = allocH(H3*HID);  W[l].cg2 = allocH(HID*HID);
    W[l].ca1 = allocH(H3*HID);  W[l].ca2 = allocH(HID*HID);
    W[l].q = allocH(HID*HC); W[l].k = allocH(HID*HC);
    W[l].v = allocH(HID*HC); W[l].e = allocH(HID*HC);
    W[l].skip = allocH(HID*HID);
    W[l].u1 = allocH(H3*H3); W[l].u2 = allocH(H3*H3);
    W[l].msg = allocH(H3*HID); W[l].mix = allocH(2*HID*HID);
  }

  auto zero = [&](float* p, size_t n) {
    k_zero<<<dim3(cdiv((int)n, 256)), 256, 0, stream>>>(p, (int)n);
  };
  auto cvt = [&](const float* Wsrc, h16* Wt, int K, int N) {
    k_cvtT<<<dim3(cdiv(K * N, 256)), 256, 0, stream>>>(Wsrc, Wt, K, N);
  };
  #define GEMM(ACT_, B0,B1,B2,O0,O1,O2,WT,BIA,MUL,OUT,M_,K_,N_) \
    k_gemm<ACT_><<<dim3(cdiv((M_),128),(N_)/128), 256, 0, stream>>>( \
        B0,B1,B2,O0,O1,O2,WT,BIA,MUL,OUT,M_,K_,N_)

  auto graphnorm = [&](float* x, const float* w, const float* b, const float* ms) {
    zero(gsum, N_GRAPHS * HID); zero(gcnt, N_GRAPHS); zero(gvar, N_GRAPHS * HID);
    k_segsum<<<dim3(cdiv(N_NODES * HID, 256)), 256, 0, stream>>>(x, batch, gsum, gcnt, N_NODES);
    k_gn_center<<<dim3(cdiv(N_NODES * HID, 256)), 256, 0, stream>>>(
        x, batch, gsum, gcnt, ms, tmpn, gvar, N_NODES);
    k_gn_final<<<dim3(cdiv(N_NODES * HID, 256)), 256, 0, stream>>>(
        tmpn, batch, gvar, gcnt, w, b, x, N_NODES);
  };

  // ---- weight convert + transpose (f32[K,N] -> f16[N,K]) ----
  cvt(rbf1w, rbf1t, HID, HID); cvt(rbf2w, rbf2t, HID, HID); cvt(fc1w, fc1t, HID, HID);
  for (int l = 0; l < 5; ++l) {
    cvt(L[l].cg1w, W[l].cg1, H3, HID);  cvt(L[l].cg2w, W[l].cg2, HID, HID);
    cvt(L[l].ca1w, W[l].ca1, H3, HID);  cvt(L[l].ca2w, W[l].ca2, HID, HID);
    cvt(L[l].qw, W[l].q, HID, HC); cvt(L[l].kw, W[l].k, HID, HC);
    cvt(L[l].vw, W[l].v, HID, HC); cvt(L[l].ew, W[l].e, HID, HC);
    cvt(L[l].skw, W[l].skip, HID, HID);
    cvt(L[l].u1w, W[l].u1, H3, H3); cvt(L[l].u2w, W[l].u2, H3, H3);
    cvt(L[l].msgw, W[l].msg, H3, HID); cvt(L[l].mixw, W[l].mix, 2*HID, HID);
  }

  // ---- constant index tables ----
  k_iota<<<dim3(cdiv(EH, 256)), 256, 0, stream>>>(offA128, EH, HID, 0);
  k_iota<<<dim3(cdiv(EH, 256)), 256, 0, stream>>>(offP0, EH, H3, 0);
  k_iota<<<dim3(cdiv(EH, 256)), 256, 0, stream>>>(offP1, EH, H3, HID);
  k_iota<<<dim3(cdiv(EH, 256)), 256, 0, stream>>>(offP2, EH, H3, 2 * HID);
  k_gath_off<<<dim3(cdiv(N_EDGES, 256)), 256, 0, stream>>>(dst, cartD, N_EDGES, HID);
  k_gath_off<<<dim3(cdiv(N_EDGES, 256)), 256, 0, stream>>>(src, cartS, N_EDGES, HID);
  k_matoff<<<dim3(cdiv(EH, 256)), 256, 0, stream>>>(dst, src, matO0, matO1, matO2);

  // ---- embedding + RBF edge features ----
  k_embed<<<dim3(cdiv(N_NODES * HID, 256)), 256, 0, stream>>>(atom_types, emb, x_c, x_m);
  k_rbf<<<dim3(cdiv(N_EDGES * HID, 256)), 256, 0, stream>>>(edge_vec, hg);
  GEMM(2, hg, nullptr, nullptr, offA128, nullptr, nullptr, rbf1t, rbf1b, nullptr, he,
       N_EDGES, HID, HID);                                   // softplus
  GEMM(0, he, nullptr, nullptr, offA128, nullptr, nullptr, rbf2t, rbf2b, nullptr, ef,
       N_EDGES, HID, HID);
  hipMemcpyAsync(e_c, ef, EDGE * sizeof(float), hipMemcpyDeviceToDevice, stream);

  for (int l = 0; l < 5; ++l) {
    const LayerP& p = L[l];
    const LayerW& w = W[l];
    // ===== CartNet branch =====
    GEMM(1, x_c, x_c, e_c, cartD, cartS, offA128, w.cg1, p.cg1b, nullptr, hg,
         N_EDGES, H3, HID);                                  // silu(cat@cg1)
    GEMM(0, hg, nullptr, nullptr, offA128, nullptr, nullptr, w.cg2, p.cg2b, nullptr, he,
         N_EDGES, HID, HID);
    GEMM(1, x_c, x_c, e_c, cartD, cartS, offA128, w.ca1, p.ca1b, nullptr, hg,
         N_EDGES, H3, HID);                                  // silu(cat@ca1) (old e_c)
    zero(st, 2 * HID);
    k_colstats<<<dim3(256), 256, 0, stream>>>(he, N_EDGES, st);
    k_bnsig<<<dim3(cdiv(N_EDGES * HID, 256)), 256, 0, stream>>>(
        he, st, p.bneg, p.bneb, sigma, e_c, N_EDGES);        // sigma + e_c += sigma
    GEMM(0, hg, nullptr, nullptr, offA128, nullptr, nullptr, w.ca2, p.ca2b, sigma, he,
         N_EDGES, HID, HID);                                 // (hg@ca2)*sigma
    zero(agg, NODE);
    k_scatter_add<<<dim3(cdiv(N_EDGES * HID, 256)), 256, 0, stream>>>(he, dst, agg, N_EDGES);
    zero(st, 2 * HID);
    k_colstats<<<dim3(256), 256, 0, stream>>>(agg, N_NODES, st);
    k_bnsilu_add<<<dim3(cdiv(N_NODES * HID, 256)), 256, 0, stream>>>(
        agg, st, p.bnng, p.bnnb, x_c, xc, N_NODES);          // xc = silu(bn(agg)) + x_c
    graphnorm(xc, p.gncw, p.gncb, p.gncm);

    // ===== Matformer branch =====
    GEMM(0, x_m, nullptr, nullptr, offA128, nullptr, nullptr, w.q, p.qb, nullptr, qb,
         N_NODES, HID, HC);
    GEMM(0, x_m, nullptr, nullptr, offA128, nullptr, nullptr, w.k, p.kb, nullptr, kb,
         N_NODES, HID, HC);
    GEMM(0, x_m, nullptr, nullptr, offA128, nullptr, nullptr, w.v, p.vb, nullptr, vb,
         N_NODES, HID, HC);
    GEMM(0, ef, nullptr, nullptr, offA128, nullptr, nullptr, w.e, p.eb, nullptr, epb,
         N_EDGES, HID, HC);
    k_gate<<<dim3(EH), 128, 0, stream>>>(qb, kb, epb, dst, src, p.lnag, p.lnab, G);
    GEMM(1, vb, vb, epb, matO0, matO1, matO2, w.u1, p.u1b, nullptr, U1,
         EH, H3, H3);                                        // silu(val_cat@upd1)
    GEMM(0, U1, U1, U1, offP0, offP1, offP2, w.u2, p.u2b, G, G,
         EH, H3, H3);                                        // (·@upd2)*gate -> G
    GEMM(0, G, G, G, offP0, offP1, offP2, w.msg, p.msgb, nullptr, mraw,
         EH, H3, HID);
    k_ln128<<<dim3(EH), 128, 0, stream>>>(mraw, p.lnmg, p.lnmb);
    zero(matt, NODE);
    k_scatter_heads<<<dim3(cdiv(EH * HID, 256)), 256, 0, stream>>>(mraw, dst, matt);
    GEMM(0, x_m, nullptr, nullptr, offA128, nullptr, nullptr, w.skip, p.skb, nullptr, x_r,
         N_NODES, HID, HID);
    k_beta_mix<<<dim3(N_NODES), 128, 0, stream>>>(matt, x_r, p.betaw, xm);
    graphnorm(xm, p.gnmw, p.gnmb, p.gnmm);

    // ===== ResidualGateMixer =====
    GEMM(3, xc, xm, nullptr, offA128, offA128, nullptr, w.mix, p.mixb, nullptr, gateN,
         N_NODES, 2 * HID, HID);                             // sigmoid([xc,xm]@mix)
    k_mix_update<<<dim3(cdiv(N_NODES * HID, 256)), 256, 0, stream>>>(gateN, xc, xm, x_c, x_m);
  }

  // ===== readout =====
  k_avg2<<<dim3(cdiv(N_NODES * HID, 256)), 256, 0, stream>>>(x_c, x_m, tmpn);
  zero(gsum, N_GRAPHS * HID); zero(gcnt, N_GRAPHS);
  k_segsum<<<dim3(cdiv(N_NODES * HID, 256)), 256, 0, stream>>>(tmpn, batch, gsum, gcnt, N_NODES);
  k_segdiv<<<dim3(cdiv(N_GRAPHS * HID, 256)), 256, 0, stream>>>(gsum, gcnt, feat);
  GEMM(1, feat, nullptr, nullptr, offA128, nullptr, nullptr, fc1t, fc1b, nullptr, feath,
       N_GRAPHS, HID, HID);                                  // silu
  k_fc2<<<dim3(N_GRAPHS), 128, 0, stream>>>(feath, fc2w, fc2b, (float*)d_out);
}
